// DSVABlockLarge_46110768889983
// MI455X (gfx1250) — matrix operand order repaired and checked by reading, not compiled, not run on hardware
//
#include <hip/hip_runtime.h>
#include <hip/hip_bf16.h>
#include <math.h>

// ---------------------------------------------------------------------------
// DSVA block for MI455X (gfx1250, wave32, WMMA).
// All GEMMs: bf16 inputs, f32 accumulation via v_wmma_f32_16x16x32_bf16.
// Weights are pre-converted to bf16 AND pre-transposed to N-major once, so
// every tile move in the GEMM hot loop is a vector b128 global/LDS op.
// ---------------------------------------------------------------------------

typedef __bf16 bf16_t;
typedef __attribute__((ext_vector_type(8)))  float  v8f;
typedef __attribute__((ext_vector_type(16))) bf16_t v16bf;
typedef __attribute__((ext_vector_type(8)))  bf16_t v8bf;

enum {
  Bc = 2, Vc = 4096, Dc = 192, Rc = 16,
  Ec = 2 * Dc,           // 384
  Hc = 8, DHc = Ec / Hc, // 48
  Kc = 16, TKc = 8,
  Mc = Bc * Vc           // 8192 token rows
};

// ------------------------------ staging ------------------------------------

__global__ void cvt_bf16_kernel(const float* __restrict__ in, bf16_t* __restrict__ out, int n) {
  int i = blockIdx.x * 256 + threadIdx.x;
  if (i < n) out[i] = (bf16_t)in[i];
}

// Wt[n*K + k] = (bf16) W[k*N + n]  (one-time weight transpose, N-major output)
__global__ void cvt_tr_bf16_kernel(const float* __restrict__ W, bf16_t* __restrict__ Wt,
                                   int K, int N) {
  int i = blockIdx.x * 256 + threadIdx.x;
  if (i >= K * N) return;
  int n = i / K, k = i - n * K;
  Wt[i] = (bf16_t)W[(size_t)k * N + n];
}

__global__ void mask_kernel(const unsigned char* __restrict__ m, float* __restrict__ f, int n) {
  int i = blockIdx.x * 256 + threadIdx.x;
  if (i < n) f[i] = m[i] ? 1.0f : 0.0f;
}

// ------------------------------ KNN ----------------------------------------
__launch_bounds__(128)
__global__ void knn_kernel(const float* __restrict__ fmask, int* __restrict__ knn) {
  int t = blockIdx.x * 128 + threadIdx.x;
  if (t >= Mc) return;
  int b = t / Vc, v = t - b * Vc;
  int ix = v >> 8, iy = (v >> 4) & 15, iz = v & 15;
  float bd[Kc];
  int   bi[Kc];
#pragma unroll
  for (int j = 0; j < Kc; ++j) { bd[j] = 3.4e38f; bi[j] = 0; }
  const float* fm = fmask + (size_t)b * Vc;
  for (int u = 0; u < Vc; ++u) {
    int ux = u >> 8, uy = (u >> 4) & 15, uz = u & 15;
    float dx = (float)(ix - ux), dy = (float)(iy - uy), dz = (float)(iz - uz);
    float d2 = (dx * dx + dy * dy + dz * dz) * 0.015625f;   // (2/16)^2 grid scale
    if (fm[u] == 0.0f) d2 += 1e9f;
    if (d2 < bd[Kc - 1]) {               // insertion keeps earlier index on ties
      int j = Kc - 1;
      while (j > 0 && bd[j - 1] > d2) { bd[j] = bd[j - 1]; bi[j] = bi[j - 1]; --j; }
      bd[j] = d2; bi[j] = u;
    }
  }
#pragma unroll
  for (int j = 0; j < Kc; ++j) knn[(size_t)t * Kc + j] = bi[j];
}

// ------------------------------ LayerNorm ----------------------------------
__launch_bounds__(256)
__global__ void ln_kernel(const float* __restrict__ x, const float* __restrict__ g,
                          const float* __restrict__ bt, bf16_t* __restrict__ out, int E) {
  int row  = blockIdx.x * 8 + (threadIdx.x >> 5);
  int lane = threadIdx.x & 31;
  const float* xr = x + (size_t)row * E;
  float s = 0.f, s2 = 0.f;
  for (int i = lane; i < E; i += 32) { float v = xr[i]; s += v; s2 += v * v; }
#pragma unroll
  for (int m = 16; m; m >>= 1) { s += __shfl_xor(s, m, 32); s2 += __shfl_xor(s2, m, 32); }
  float mean = s / (float)E;
  float var  = s2 / (float)E - mean * mean;
  float r    = rsqrtf(var + 1e-5f);
  bf16_t* orow = out + (size_t)row * E;
  for (int i = lane; i < E; i += 32)
    orow[i] = (bf16_t)((xr[i] - mean) * r * g[i] + bt[i]);
}

// ------------------------------ WMMA GEMM ----------------------------------
// C[M,N] = epi( A[M,K](bf16, K-major) @ Bt[N,K](bf16, N-major = W^T) + bias )
// Block: 128 threads = 4 waves; block tile 128(M) x 64(N); wave tile 64x32;
// per K-step each wave: 2 resident B fragments, 4 A fragments, 8 WMMAs.
#define LDA 40   // 80 B row stride: 16B-aligned fragments, bank-spread
#define LDB 48   // 96 B row stride: 32B-aligned fragments

__device__ __forceinline__ float gelu_tanh_fast(float x) {
  // tanh-approx gelu via sigmoid: x * sigmoid(2*0.79788456*(x+0.044715x^3))
  float z = x + 0.044715f * x * x * x;
  return x / (1.0f + __builtin_amdgcn_exp2f(-2.3022077f * z));  // 2*0.79788456*log2(e)
}

template <bool GELU, bool ROWMASK, bool RESID, bool WF, bool WB>
__launch_bounds__(128)
__global__ void gemm_bf16_wmma(const bf16_t* __restrict__ A, const bf16_t* __restrict__ Bt,
                               const float* __restrict__ bias,
                               float* __restrict__ Cf, bf16_t* __restrict__ Cb,
                               const float* __restrict__ resid,
                               const float* __restrict__ rowmask,
                               float scale, int M, int N, int Kd) {
  __shared__ __attribute__((aligned(32))) bf16_t sA[128 * LDA];
  __shared__ __attribute__((aligned(32))) bf16_t sB[64 * LDB];

  const int tid  = threadIdx.x;
  const int lane = tid & 31, wave = tid >> 5;
  const int wm   = (wave >> 1) * 64, wn = (wave & 1) * 32;
  const int m0   = blockIdx.y * 128, n0 = blockIdx.x * 64;

  v8f acc[4][2] = {};

  const int fmrow = lane & 15;        // row/col within a 16x16 tile
  const int kbA   = (lane >> 4) * 16; // A K-group (contiguous, same as B)
  const int kgB   = (lane >> 4) * 16; // B K-group (contiguous, same as A; N-major LDS)

  const int ar = tid >> 2, ac = (tid & 3) * 8;   // A-tile loader coords

  for (int k0 = 0; k0 < Kd; k0 += 32) {
    __syncthreads();
    // A tile: 128 rows x 32 k  -> 4x v8bf per thread
#pragma unroll
    for (int c = 0; c < 4; ++c) {
      int r = ar + c * 32;
      v8bf av = *(const v8bf*)(A + (size_t)(m0 + r) * Kd + k0 + ac);
      *(v8bf*)(&sA[r * LDA + ac]) = av;
    }
    // B tile: 64 n-rows x 32 k (already N-major in global) -> 2x v8bf per thread
#pragma unroll
    for (int c = 0; c < 2; ++c) {
      int r = (tid >> 2) + c * 32;   // n within tile
      v8bf bv = *(const v8bf*)(Bt + (size_t)(n0 + r) * Kd + k0 + ac);
      *(v8bf*)(&sB[r * LDB + ac]) = bv;
    }
    // prefetch next K tile into L2/WGP$ (global_prefetch_b8)
    if (k0 + 32 < Kd) {
      __builtin_prefetch(A  + (size_t)(m0 + ar) * Kd + k0 + 32, 0, 1);
      __builtin_prefetch(Bt + (size_t)(n0 + (tid >> 2)) * Kd + k0 + 32, 0, 1);
    }
    __syncthreads();

    // 2 resident B fragments
    v16bf bf0 = *(const v16bf*)(&sB[(wn +  0 + fmrow) * LDB + kgB]);
    v16bf bf1 = *(const v16bf*)(&sB[(wn + 16 + fmrow) * LDB + kgB]);
#pragma unroll
    for (int si = 0; si < 4; ++si) {
      const bf16_t* pa = &sA[(wm + si * 16 + fmrow) * LDA + kbA];
      v8bf alo = *(const v8bf*)(pa);        // K = kb .. kb+7
      v8bf ahi = *(const v8bf*)(pa + 8);    // K = kb+8 .. kb+15
      v16bf a  = __builtin_shufflevector(alo, ahi,
                   0,1,2,3,4,5,6,7,8,9,10,11,12,13,14,15);
      acc[si][0] = __builtin_amdgcn_wmma_f32_16x16x32_bf16(
          false, a, false, bf0, (short)0, acc[si][0], false, false);
      acc[si][1] = __builtin_amdgcn_wmma_f32_16x16x32_bf16(
          false, a, false, bf1, (short)0, acc[si][1], false, false);
    }
  }

  // Epilogue: C/D layout -> lane owns col (lane&15), rows (lane>>4)*8 + r
#pragma unroll
  for (int si = 0; si < 4; ++si) {
#pragma unroll
    for (int sj = 0; sj < 2; ++sj) {
      int gm = m0 + wm + si * 16 + (lane >> 4) * 8;
      int gn = n0 + wn + sj * 16 + (lane & 15);
      float bv = bias[gn];
#pragma unroll
      for (int r = 0; r < 8; ++r) {
        int m   = gm + r;
        float t = acc[si][sj][r] + bv;
        if (GELU)    t = gelu_tanh_fast(t);
        if (ROWMASK) t *= rowmask[m];
        t *= scale;
        size_t idx = (size_t)m * N + gn;
        if (RESID)   t += resid[idx];
        if (WF)      Cf[idx] = t;
        if (WB)      Cb[idx] = (bf16_t)t;
      }
    }
  }
}

// ------------------------------ sparse attention ---------------------------
__launch_bounds__(128)
__global__ void attn_kernel(const float* __restrict__ qkv, const int* __restrict__ knn,
                            const float* __restrict__ fmask, bf16_t* __restrict__ ob) {
  int t = blockIdx.x * 128 + threadIdx.x;
  if (t >= Mc * Hc) return;
  int h   = t & (Hc - 1);
  int row = t >> 3;               // row in [0, B*V)
  int b   = row / Vc;
  const int QW = 3 * Ec;          // 1152

  const size_t qb = (size_t)row * QW + h * DHc;
  float q[DHc];
#pragma unroll
  for (int d = 0; d < DHc; ++d) q[d] = qkv[qb + d];

  float sc[Kc];
#pragma unroll 1
  for (int kk = 0; kk < Kc; ++kk) {
    int u = knn[(size_t)row * Kc + kk];
    size_t kb = ((size_t)(b * Vc + u)) * QW + Ec + h * DHc;
    float s = 0.f;
#pragma unroll
    for (int d = 0; d < DHc; ++d) s += q[d] * qkv[kb + d];
    s *= 0.14433756729740643f;    // 1/sqrt(48)
    if (fmask[b * Vc + u] == 0.0f) s = -1e9f;
    sc[kk] = s;
  }

  float ssel[TKc]; int sel[TKc]; bool used[Kc];
#pragma unroll
  for (int kk = 0; kk < Kc; ++kk) used[kk] = false;
#pragma unroll
  for (int j = 0; j < TKc; ++j) {
    float best = -3.4e38f; int bi = 0;
#pragma unroll
    for (int kk = 0; kk < Kc; ++kk)
      if (!used[kk] && sc[kk] > best) { best = sc[kk]; bi = kk; }
    used[bi] = true; ssel[j] = best; sel[j] = bi;
  }

  float mx = ssel[0], den = 0.f, w[TKc];
#pragma unroll
  for (int j = 0; j < TKc; ++j) { w[j] = __builtin_amdgcn_exp2f(1.4426950408889634f * (ssel[j] - mx)); den += w[j]; }
  float inv = 1.0f / den;

  float accv[DHc];
#pragma unroll
  for (int d = 0; d < DHc; ++d) accv[d] = 0.f;
#pragma unroll 1
  for (int j = 0; j < TKc; ++j) {
    int u = knn[(size_t)row * Kc + sel[j]];
    size_t vb = ((size_t)(b * Vc + u)) * QW + 2 * Ec + h * DHc;
    float wj = w[j] * inv;
#pragma unroll
    for (int d = 0; d < DHc; ++d) accv[d] += wj * qkv[vb + d];
  }

  size_t obase = (size_t)row * Ec + h * DHc;
#pragma unroll
  for (int d = 0; d < DHc; ++d) ob[obase + d] = (bf16_t)accv[d];
}

// ------------------------------ launcher -----------------------------------

template <bool GELU, bool ROWMASK, bool RESID, bool WF, bool WB>
static void launch_gemm(const bf16_t* A, const bf16_t* Bt, const float* bias,
                        float* Cf, bf16_t* Cb, const float* resid, const float* rowm,
                        float scale, int M, int N, int K, hipStream_t stream) {
  dim3 g((unsigned)(N / 64), (unsigned)(M / 128));
  gemm_bf16_wmma<GELU, ROWMASK, RESID, WF, WB>
      <<<g, 128, 0, stream>>>(A, Bt, bias, Cf, Cb, resid, rowm, scale, M, N, K);
}

extern "C" void kernel_launch(void* const* d_in, const int* in_sizes, int n_in,
                              void* d_out, int out_size, void* d_ws, size_t ws_size,
                              hipStream_t stream) {
  (void)in_sizes; (void)n_in; (void)out_size; (void)ws_size;

  const float*         voxel = (const float*)d_in[0];
  const unsigned char* nem   = (const unsigned char*)d_in[1];
  const float* We   = (const float*)d_in[2];
  const float* be   = (const float*)d_in[3];
  const float* g1   = (const float*)d_in[4];
  const float* bt1  = (const float*)d_in[5];
  const float* Wqkv = (const float*)d_in[6];
  const float* bqkv = (const float*)d_in[7];
  const float* Wo   = (const float*)d_in[8];
  const float* bo   = (const float*)d_in[9];
  const float* Wc   = (const float*)d_in[10];
  const float* bc   = (const float*)d_in[11];
  const float* g2   = (const float*)d_in[12];
  const float* bt2  = (const float*)d_in[13];
  const float* W1   = (const float*)d_in[14];
  const float* b1   = (const float*)d_in[15];
  const float* W2   = (const float*)d_in[16];
  const float* b2   = (const float*)d_in[17];

  char* ws = (char*)d_ws;
  size_t off = 0;
  auto alloc = [&](size_t bytes) -> char* {
    char* p = ws + off;
    off += (bytes + 255) & ~(size_t)255;
    return p;
  };

  // bf16 staging (weights transposed to N-major)
  bf16_t* Web   = (bf16_t*)alloc((size_t)Dc * Ec * 2);
  bf16_t* Wqkvb = (bf16_t*)alloc((size_t)Ec * 3 * Ec * 2);
  bf16_t* Wob   = (bf16_t*)alloc((size_t)Ec * Ec * 2);
  bf16_t* Wcb   = (bf16_t*)alloc((size_t)Ec * Dc * 2);
  bf16_t* W1b   = (bf16_t*)alloc((size_t)Ec * 2 * Ec * 2);
  bf16_t* W2b   = (bf16_t*)alloc((size_t)2 * Ec * Ec * 2);
  bf16_t* Xb    = (bf16_t*)alloc((size_t)Mc * Dc * 2);
  float*  fmaskf= (float*) alloc((size_t)Mc * 4);
  int*    knnb  = (int*)   alloc((size_t)Mc * Kc * 4);
  float*  x1f   = (float*) alloc((size_t)Mc * Ec * 4);   // reused as x2f
  bf16_t* x1nb  = (bf16_t*)alloc((size_t)Mc * Ec * 2);   // reused as x2nb
  float*  qkvf  = (float*) alloc((size_t)Mc * 3 * Ec * 4);
  bf16_t* ob    = (bf16_t*)alloc((size_t)Mc * Ec * 2);
  bf16_t* o2b   = (bf16_t*)alloc((size_t)Mc * Ec * 2);
  float*  xresf = (float*) alloc((size_t)Mc * Dc * 4);
  bf16_t* xresb = (bf16_t*)alloc((size_t)Mc * Dc * 2);
  // h1/h2 reuse the (dead after attention) qkv buffer: 12.6MB + 6.3MB < 37.7MB
  bf16_t* h1b = (bf16_t*)qkvf;
  bf16_t* h2b = (bf16_t*)((char*)qkvf + ((size_t)16 << 20));

  auto cvtT = [&](const float* src, bf16_t* dst, int K, int N) {   // -> N-major
    int n = K * N;
    cvt_tr_bf16_kernel<<<(n + 255) / 256, 256, 0, stream>>>(src, dst, K, N);
  };

  // Stage 0: precision staging + mask + KNN
  cvtT(We,   Web,   Dc, Ec);
  cvtT(Wqkv, Wqkvb, Ec, 3 * Ec);
  cvtT(Wo,   Wob,   Ec, Ec);
  cvtT(Wc,   Wcb,   Ec, Dc);
  cvtT(W1,   W1b,   Ec, 2 * Ec);
  cvtT(W2,   W2b,   2 * Ec, Ec);
  cvt_bf16_kernel<<<(Mc * Dc + 255) / 256, 256, 0, stream>>>(voxel, Xb, Mc * Dc);
  mask_kernel<<<(Mc + 255) / 256, 256, 0, stream>>>(nem, fmaskf, Mc);
  knn_kernel<<<Mc / 128, 128, 0, stream>>>(fmaskf, knnb);

  // x1 = X @ We + be ; LN(g1,bt1)
  launch_gemm<false,false,false,true,false>(Xb, Web, be, x1f, nullptr, nullptr, nullptr,
                                            1.0f, Mc, Ec, Dc, stream);
  ln_kernel<<<Mc / 8, 256, 0, stream>>>(x1f, g1, bt1, x1nb, Ec);

  // qkv = x1n @ Wqkv + bqkv
  launch_gemm<false,false,false,true,false>(x1nb, Wqkvb, bqkv, qkvf, nullptr, nullptr, nullptr,
                                            1.0f, Mc, 3 * Ec, Ec, stream);

  // sparse top-k attention -> o (bf16)
  attn_kernel<<<(Mc * Hc) / 128, 128, 0, stream>>>(qkvf, knnb, fmaskf, ob);

  // o2 = (o @ Wo + bo) * fmask   (masked scatter)
  launch_gemm<false,true,false,false,true>(ob, Wob, bo, nullptr, o2b, nullptr, fmaskf,
                                           1.0f, Mc, Ec, Ec, stream);

  // xres = 0.5*(o2 @ Wc + bc) + shortcut
  launch_gemm<false,false,true,true,true>(o2b, Wcb, bc, xresf, xresb, voxel, nullptr,
                                          0.5f, Mc, Dc, Ec, stream);

  // x2 = xres @ We + be ; LN(g2,bt2)   (reuse x1f/x1nb)
  launch_gemm<false,false,false,true,false>(xresb, Web, be, x1f, nullptr, nullptr, nullptr,
                                            1.0f, Mc, Ec, Dc, stream);
  ln_kernel<<<Mc / 8, 256, 0, stream>>>(x1f, g2, bt2, x1nb, Ec);

  // FFN: h1 = gelu(x2n @ W1 + b1) ; h2 = h1 @ W2 + b2
  launch_gemm<true,false,false,false,true>(x1nb, W1b, b1, nullptr, h1b, nullptr, nullptr,
                                           1.0f, Mc, 2 * Ec, Ec, stream);
  launch_gemm<false,false,false,false,true>(h1b, W2b, b2, nullptr, h2b, nullptr, nullptr,
                                            1.0f, Mc, Ec, 2 * Ec, stream);

  // out = 0.5*(h2 @ Wc + bc) + xres
  launch_gemm<false,false,true,true,false>(h2b, Wcb, bc, (float*)d_out, nullptr, xresf, nullptr,
                                           0.5f, Mc, Dc, Ec, stream);
}
